// LeadTransformer_56822417326630
// MI455X (gfx1250) — matrix operand order repaired and checked by reading, not compiled, not run on hardware
//
#include <hip/hip_runtime.h>
#include <hip/hip_bf16.h>

// ---------------------------------------------------------------------------
// Types / helpers
// ---------------------------------------------------------------------------
typedef __attribute__((ext_vector_type(16))) __bf16 v16bf;
typedef __attribute__((ext_vector_type(8)))  float  v8f;
typedef __attribute__((ext_vector_type(4)))  unsigned int u32x4;
typedef __attribute__((ext_vector_type(8)))  int i32x8;
typedef __attribute__((ext_vector_type(4)))  int i32x4;

union Frag16 {
  v16bf v;
  unsigned int  w[8];
  unsigned short u[16];
};

__device__ __forceinline__ unsigned short f2bf(float f) {
  unsigned int u = __float_as_uint(f);
  u += 0x7FFFu + ((u >> 16) & 1u);          // round-to-nearest-even
  return (unsigned short)(u >> 16);
}

// ---------------------------------------------------------------------------
// Tensor Data Mover: 2D bf16 tile load (global -> LDS), linear LDS fill.
// tile_x elements per row (contiguous), tile_y rows, row stride in elements.
// Issued by one wave; completion via s_wait_tensorcnt.
// ---------------------------------------------------------------------------
__device__ __forceinline__ void tdm_load_2d_bf16(const unsigned short* lds_dst,
                                                 const unsigned short* gsrc,
                                                 int tile_x, int tile_y,
                                                 int stride_elems) {
  unsigned long long ga = (unsigned long long)(const void*)gsrc;
  unsigned int lds_off = (unsigned int)(unsigned long long)(const void*)lds_dst;
  u32x4 g0;
  g0[0] = 1u;                                        // count=1 (valid user D#)
  g0[1] = lds_off;                                   // lds_addr (bytes)
  g0[2] = (unsigned int)ga;                          // global_addr[31:0]
  g0[3] = (unsigned int)((ga >> 32) & 0x01FFFFFFu)   // global_addr[56:32]
          | 0x80000000u;                             // type=2 ("image")
  i32x8 g1;
  g1[0] = (int)(1u << 16);                           // data_size=1 -> 2 bytes
  g1[1] = 0;                                         // tensor_dim0 lo16 (dim=1<<20)
  g1[2] = 0x0010;                                    // tensor_dim0 hi16, dim1 lo16=0
  g1[3] = 0x0010 | (tile_x << 16);                   // tensor_dim1 hi16, tile_dim0
  g1[4] = tile_y;                                    // tile_dim1, tile_dim2=0
  g1[5] = stride_elems;                              // tensor_dim0_stride[31:0]
  g1[6] = 0;                                         // stride hi, dim1_stride lo
  g1[7] = 0;                                         // dim1_stride hi
  i32x4 z4 = {};
  i32x8 z8 = {};
  __builtin_amdgcn_tensor_load_to_lds(g0, g1, z4, z4, z8, 0);
}

// ---------------------------------------------------------------------------
// f32 -> bf16 conversion (weights)
// ---------------------------------------------------------------------------
__global__ __launch_bounds__(256)
void cvt_bf16_kernel(const float* __restrict__ in, unsigned short* __restrict__ out, int n) {
  for (int i = blockIdx.x * 256 + threadIdx.x; i < n; i += gridDim.x * 256)
    out[i] = f2bf(in[i]);
}

// ---------------------------------------------------------------------------
// LayerNorm over D=256, one wave (32 lanes) per row, 8 rows per block
// ---------------------------------------------------------------------------
template<bool BF16OUT>
__global__ __launch_bounds__(256)
void ln_kernel(const float* __restrict__ x, const float* __restrict__ g,
               const float* __restrict__ bt, unsigned short* __restrict__ outh,
               float* __restrict__ outf) {
  const int lane = threadIdx.x & 31;
  const int wv = threadIdx.x >> 5;
  const size_t row = (size_t)blockIdx.x * 8 + wv;
  const size_t base = row * 256;

  float vals[8];
  float sum = 0.f;
#pragma unroll
  for (int j = 0; j < 8; ++j) { vals[j] = x[base + lane + j * 32]; sum += vals[j]; }
#pragma unroll
  for (int off = 16; off > 0; off >>= 1) sum += __shfl_xor(sum, off, 32);
  const float mean = sum * (1.0f / 256.0f);

  float var = 0.f;
#pragma unroll
  for (int j = 0; j < 8; ++j) { float d = vals[j] - mean; var += d * d; }
#pragma unroll
  for (int off = 16; off > 0; off >>= 1) var += __shfl_xor(var, off, 32);
  const float rstd = rsqrtf(var * (1.0f / 256.0f) + 1e-5f);

#pragma unroll
  for (int j = 0; j < 8; ++j) {
    int col = lane + j * 32;
    float y = (vals[j] - mean) * rstd * g[col] + bt[col];
    if (BF16OUT) outh[base + col] = f2bf(y);
    else         outf[base + col] = y;
  }
}

// ---------------------------------------------------------------------------
// bf16 GEMM: C[M,N] = A[M,K] * B[K,N] + bias, fused epilogue
//   EPI 0: write bf16      EPI 1: GELU -> bf16      EPI 2: +residual -> f32
// 64x64 block tile, 4 waves, each wave 16 rows x 64 cols.
// K stepped by 64: A tile staged by TDM, B tile K-transposed manually.
// ---------------------------------------------------------------------------
template<int EPI>
__global__ __launch_bounds__(128)
void gemm_bf16_kernel(const unsigned short* __restrict__ A,
                      const unsigned short* __restrict__ B,
                      const float* __restrict__ bias,
                      const float* __restrict__ res,
                      float* __restrict__ outf,
                      unsigned short* __restrict__ outh,
                      int M, int N, int K) {
  __shared__ __align__(16) unsigned short As[64 * 64];   // [m][k]
  __shared__ __align__(16) unsigned short Bt[64 * 64];   // [n][k] (K-transposed)

  const int tid = threadIdx.x;
  const int lane = tid & 31, wv = tid >> 5;
  const int hlf = lane >> 4, l16 = lane & 15;
  const int m0 = blockIdx.y * 64, n0 = blockIdx.x * 64;

  v8f acc[4] = {};

  for (int k0 = 0; k0 < K; k0 += 64) {
    if (wv == 0)                                   // TDM: A tile 64x64, stride K
      tdm_load_2d_bf16(As, &A[(size_t)m0 * K + k0], 64, 64, K);
#pragma unroll
    for (int it = 0; it < 4; ++it) {               // B tile 64x64, transposed into LDS
      int li = tid + it * 128;
      int kr = li >> 3, nc = (li & 7) << 3;
      uint4 val = *reinterpret_cast<const uint4*>(&B[(size_t)(k0 + kr) * N + n0 + nc]);
      const unsigned short* u = reinterpret_cast<const unsigned short*>(&val);
#pragma unroll
      for (int e = 0; e < 8; ++e) Bt[(nc + e) * 64 + kr] = u[e];
    }
    if (wv == 0) __builtin_amdgcn_s_wait_tensorcnt(0);
    __syncthreads();

#pragma unroll
    for (int ch = 0; ch < 2; ++ch) {
      Frag16 a;
#pragma unroll
      for (int v = 0; v < 8; ++v) {
        int k = (ch << 5) + ((v & 3) << 1) + ((v >> 2) << 4) + (hlf << 3);
        a.w[v] = *reinterpret_cast<const unsigned int*>(&As[((wv << 4) + l16) * 64 + k]);
      }
#pragma unroll
      for (int t = 0; t < 4; ++t) {
        Frag16 b;
#pragma unroll
        for (int v = 0; v < 8; ++v)
          b.w[v] = *reinterpret_cast<const unsigned int*>(
              &Bt[((t << 4) + l16) * 64 + (ch << 5) + (hlf << 3) + ((v & 3) << 1) + ((v >> 2) << 4)]);
        acc[t] = __builtin_amdgcn_wmma_f32_16x16x32_bf16(false, a.v, false, b.v,
                                                         (short)0, acc[t], false, false);
      }
    }
    __syncthreads();
  }

#pragma unroll
  for (int t = 0; t < 4; ++t) {
#pragma unroll
    for (int v = 0; v < 8; ++v) {
      int m = m0 + (wv << 4) + (hlf << 3) + v;
      int n = n0 + (t << 4) + l16;
      float val = acc[t][v] + bias[n];
      if (EPI == 1) {                              // tanh-approx GELU (jax default)
        float xx = val;
        float c = xx + 0.044715f * xx * xx * xx;
        val = 0.5f * xx * (1.0f + tanhf(0.7978845608028654f * c));
      }
      if (EPI == 2) {
        val += res[(size_t)m * N + n];
        outf[(size_t)m * N + n] = val;
      } else {
        outh[(size_t)m * N + n] = f2bf(val);
      }
    }
  }
}

// ---------------------------------------------------------------------------
// Flash attention: one WG per (seq, head, 64-query block). 4 waves.
// qkv: bf16 [96*512, 768] (q|k|v each 256 = 4 heads * 64)
// rel: f32 [1023, 4] (already offset to layer d)
// o  : bf16 [96*512, 256]
// Q and K tiles staged via TDM; V transposed manually.
// ---------------------------------------------------------------------------
__global__ __launch_bounds__(128)
void attn_kernel(const unsigned short* __restrict__ qkv,
                 const float* __restrict__ rel,
                 unsigned short* __restrict__ o) {
  __shared__ __align__(16) unsigned short Qs[64 * 64];     // [q][dh]
  __shared__ __align__(16) unsigned short Ks[64 * 64];     // [key][dh]
  __shared__ __align__(16) unsigned short Vt[64 * 64];     // [dh][key]
  __shared__ __align__(16) unsigned short Ps[4 * 16 * 64]; // per-wave P [16][64]

  const int tid = threadIdx.x;
  const int lane = tid & 31, wv = tid >> 5;
  const int hlf = lane >> 4, l16 = lane & 15;
  const int q0 = blockIdx.x * 64;
  const int head = blockIdx.y;
  const size_t rowbase = (size_t)blockIdx.z * 512;
  const float scale = 0.125f;  // 64^-0.5

  if (wv == 0)                                   // TDM: Q tile, stride 768
    tdm_load_2d_bf16(Qs, &qkv[(rowbase + q0) * 768 + head * 64], 64, 64, 768);

  float m_i[8], l_i[8];
  v8f Oacc[4] = {};
#pragma unroll
  for (int v = 0; v < 8; ++v) { m_i[v] = -3.0e38f; l_i[v] = 0.f; }

  for (int kb = 0; kb < 8; ++kb) {
    __syncthreads();                             // previous-iter LDS consumers done
    const int kbase = kb * 64;
    if (wv == 0)                                 // TDM: K tile, stride 768
      tdm_load_2d_bf16(Ks, &qkv[(rowbase + kbase) * 768 + 256 + head * 64], 64, 64, 768);
#pragma unroll
    for (int it = 0; it < 4; ++it) {             // stage V transposed
      int li = tid + it * 128;
      int r = li >> 3, c = (li & 7) << 3;
      uint4 vvec = *reinterpret_cast<const uint4*>(
          &qkv[(rowbase + kbase + r) * 768 + 512 + head * 64 + c]);
      const unsigned short* u = reinterpret_cast<const unsigned short*>(&vvec);
#pragma unroll
      for (int e = 0; e < 8; ++e) Vt[(c + e) * 64 + r] = u[e];
    }
    if (wv == 0) __builtin_amdgcn_s_wait_tensorcnt(0);  // drains Q (kb=0) + K
    __syncthreads();

    // S = Q * K^T  (16 q-rows per wave x 64 keys)
    v8f S[4] = {};
#pragma unroll
    for (int ch = 0; ch < 2; ++ch) {
      Frag16 a;
#pragma unroll
      for (int v = 0; v < 8; ++v) {
        int k = (ch << 5) + ((v & 3) << 1) + ((v >> 2) << 4) + (hlf << 3);
        a.w[v] = *reinterpret_cast<const unsigned int*>(&Qs[((wv << 4) + l16) * 64 + k]);
      }
#pragma unroll
      for (int t = 0; t < 4; ++t) {
        Frag16 b;
#pragma unroll
        for (int v = 0; v < 8; ++v)
          b.w[v] = *reinterpret_cast<const unsigned int*>(
              &Ks[((t << 4) + l16) * 64 + (ch << 5) + (hlf << 3) + ((v & 3) << 1) + ((v >> 2) << 4)]);
        S[t] = __builtin_amdgcn_wmma_f32_16x16x32_bf16(false, a.v, false, b.v,
                                                       (short)0, S[t], false, false);
      }
    }

    // scale + rel-pos bias + online softmax
#pragma unroll
    for (int v = 0; v < 8; ++v) {
      int i = q0 + (wv << 4) + (hlf << 3) + v;
      float mx = -3.0e38f;
#pragma unroll
      for (int t = 0; t < 4; ++t) {
        int j = kbase + (t << 4) + l16;
        float s = S[t][v] * scale + rel[(i - j + 511) * 4 + head];
        S[t][v] = s;
        mx = fmaxf(mx, s);
      }
      mx = fmaxf(mx, __shfl_xor(mx, 1, 32));
      mx = fmaxf(mx, __shfl_xor(mx, 2, 32));
      mx = fmaxf(mx, __shfl_xor(mx, 4, 32));
      mx = fmaxf(mx, __shfl_xor(mx, 8, 32));
      float mnew = fmaxf(m_i[v], mx);
      float al = __expf(m_i[v] - mnew);
      m_i[v] = mnew;
      float rs = 0.f;
#pragma unroll
      for (int t = 0; t < 4; ++t) {
        float p = __expf(S[t][v] - mnew);
        S[t][v] = p;
        rs += p;
      }
      rs += __shfl_xor(rs, 1, 32);
      rs += __shfl_xor(rs, 2, 32);
      rs += __shfl_xor(rs, 4, 32);
      rs += __shfl_xor(rs, 8, 32);
      l_i[v] = l_i[v] * al + rs;
#pragma unroll
      for (int t = 0; t < 4; ++t) {
        Oacc[t][v] *= al;
        Ps[(wv << 10) + (((hlf << 3) + v) << 6) + (t << 4) + l16] = f2bf(S[t][v]);
      }
    }

    // O += P * V
#pragma unroll
    for (int ch = 0; ch < 2; ++ch) {
      Frag16 a;
#pragma unroll
      for (int v = 0; v < 8; ++v) {
        int k = (ch << 5) + ((v & 3) << 1) + ((v >> 2) << 4) + (hlf << 3);
        a.w[v] = *reinterpret_cast<const unsigned int*>(&Ps[(wv << 10) + (l16 << 6) + k]);
      }
#pragma unroll
      for (int t = 0; t < 4; ++t) {
        Frag16 b;
#pragma unroll
        for (int v = 0; v < 8; ++v)
          b.w[v] = *reinterpret_cast<const unsigned int*>(
              &Vt[((t << 4) + l16) * 64 + (ch << 5) + (hlf << 3) + ((v & 3) << 1) + ((v >> 2) << 4)]);
        Oacc[t] = __builtin_amdgcn_wmma_f32_16x16x32_bf16(false, a.v, false, b.v,
                                                          (short)0, Oacc[t], false, false);
      }
    }
  }

#pragma unroll
  for (int t = 0; t < 4; ++t) {
#pragma unroll
    for (int v = 0; v < 8; ++v) {
      int i = q0 + (wv << 4) + (hlf << 3) + v;
      float val = Oacc[t][v] / l_i[v];
      o[(rowbase + i) * 256 + head * 64 + (t << 4) + l16] = f2bf(val);
    }
  }
}

// ---------------------------------------------------------------------------
// Host-side orchestration
// ---------------------------------------------------------------------------
extern "C" void kernel_launch(void* const* d_in, const int* in_sizes, int n_in,
                              void* d_out, int out_size, void* d_ws, size_t ws_size,
                              hipStream_t stream) {
  (void)in_sizes; (void)n_in; (void)out_size; (void)ws_size;
  const float* x      = (const float*)d_in[0];
  const float* ln1_s  = (const float*)d_in[1];
  const float* ln1_b  = (const float*)d_in[2];
  const float* qkv_w  = (const float*)d_in[3];
  const float* qkv_b  = (const float*)d_in[4];
  const float* proj_w = (const float*)d_in[5];
  const float* proj_b = (const float*)d_in[6];
  const float* rel    = (const float*)d_in[7];
  const float* ln2_s  = (const float*)d_in[8];
  const float* ln2_b  = (const float*)d_in[9];
  const float* w1     = (const float*)d_in[10];
  const float* b1     = (const float*)d_in[11];
  const float* w2     = (const float*)d_in[12];
  const float* b2     = (const float*)d_in[13];
  const float* lnf_s  = (const float*)d_in[14];
  const float* lnf_b  = (const float*)d_in[15];

  const size_t M = 49152;  // 8*12*512
  char* ws = (char*)d_ws;
  size_t off = 0;
  auto alloc = [&](size_t bytes) -> void* {
    void* p = (void*)(ws + off);
    off += (bytes + 255) & ~(size_t)255;
    return p;
  };

  float*          h   = (float*)alloc(M * 256 * 4);
  unsigned short* xn  = (unsigned short*)alloc(M * 256 * 2);
  unsigned short* qkv = (unsigned short*)alloc(M * 768 * 2);
  unsigned short* ob  = (unsigned short*)alloc(M * 256 * 2);
  unsigned short* a1  = (unsigned short*)alloc(M * 1024 * 2);
  unsigned short* wqb = (unsigned short*)alloc((size_t)6 * 256 * 768 * 2);
  unsigned short* wpb = (unsigned short*)alloc((size_t)6 * 256 * 256 * 2);
  unsigned short* w1b = (unsigned short*)alloc((size_t)6 * 256 * 1024 * 2);
  unsigned short* w2b = (unsigned short*)alloc((size_t)6 * 1024 * 256 * 2);

  // weights -> bf16 (deterministic, every call)
  cvt_bf16_kernel<<<2048, 256, 0, stream>>>(qkv_w, wqb, 6 * 256 * 768);
  cvt_bf16_kernel<<<2048, 256, 0, stream>>>(proj_w, wpb, 6 * 256 * 256);
  cvt_bf16_kernel<<<2048, 256, 0, stream>>>(w1, w1b, 6 * 256 * 1024);
  cvt_bf16_kernel<<<2048, 256, 0, stream>>>(w2, w2b, 6 * 1024 * 256);

  (void)hipMemcpyAsync(h, x, M * 256 * sizeof(float), hipMemcpyDeviceToDevice, stream);

  const dim3 lnGrid(M / 8), lnBlk(256);
  const dim3 gemmBlk(128);
  const dim3 attnGrid(8, 4, 96), attnBlk(128);

  for (int d = 0; d < 6; ++d) {
    // --- attention block ---
    ln_kernel<true><<<lnGrid, lnBlk, 0, stream>>>(h, ln1_s + d * 256, ln1_b + d * 256, xn, nullptr);
    gemm_bf16_kernel<0><<<dim3(768 / 64, M / 64), gemmBlk, 0, stream>>>(
        xn, wqb + (size_t)d * 256 * 768, qkv_b + d * 768, nullptr, nullptr, qkv,
        (int)M, 768, 256);
    attn_kernel<<<attnGrid, attnBlk, 0, stream>>>(qkv, rel + (size_t)d * 1023 * 4, ob);
    gemm_bf16_kernel<2><<<dim3(256 / 64, M / 64), gemmBlk, 0, stream>>>(
        ob, wpb + (size_t)d * 256 * 256, proj_b + d * 256, h, h, nullptr,
        (int)M, 256, 256);
    // --- FFN block ---
    ln_kernel<true><<<lnGrid, lnBlk, 0, stream>>>(h, ln2_s + d * 256, ln2_b + d * 256, xn, nullptr);
    gemm_bf16_kernel<1><<<dim3(1024 / 64, M / 64), gemmBlk, 0, stream>>>(
        xn, w1b + (size_t)d * 256 * 1024, b1 + d * 1024, nullptr, nullptr, a1,
        (int)M, 1024, 256);
    gemm_bf16_kernel<2><<<dim3(256 / 64, M / 64), gemmBlk, 0, stream>>>(
        a1, w2b + (size_t)d * 1024 * 256, b2 + d * 256, h, h, nullptr,
        (int)M, 256, 1024);
  }

  ln_kernel<false><<<lnGrid, lnBlk, 0, stream>>>(h, lnf_s, lnf_b, nullptr, (float*)d_out);
}
